// FixedFMoE_36679020708560
// MI455X (gfx1250) — matrix operand order repaired and checked by reading, not compiled, not run on hardware
//
#include <hip/hip_runtime.h>
#include <hip/hip_bf16.h>
#include <math.h>

// ---------------- problem dims ----------------
#define B_TOK  2048
#define DMODEL 1024
#define HDIM   4096
#define NEXP   8
#define TOPK   2
#define NSLOT  (B_TOK * TOPK)   // 4096 slots total (each token -> exactly 2 experts)

// ---------------- vector types ----------------
typedef __attribute__((ext_vector_type(16))) __bf16 v16bf;
typedef __attribute__((ext_vector_type(8)))  __bf16 v8bf;
typedef __attribute__((ext_vector_type(4)))  __bf16 v4bf;
typedef __attribute__((ext_vector_type(8)))  float  v8f;
typedef __attribute__((ext_vector_type(4)))  float  v4f;
typedef int v2i_gcc __attribute__((vector_size(8)));   // matches async builtin param

// ---------------- workspace layout (bytes) ----------------
#define WS_COUNTS   ((size_t)0)                    // NEXP int
#define WS_CURSORS  ((size_t)64)                   // NEXP int
#define WS_OFFSETS  ((size_t)128)                  // NEXP int
#define WS_SEL_E    ((size_t)4096)                 // NSLOT int
#define WS_SEL_G    ((size_t)(4096 + NSLOT * 4))   // NSLOT float
#define WS_SLOT_OF  ((size_t)(4096 + NSLOT * 8))   // NSLOT int
#define WS_TOKEN_OF ((size_t)(4096 + NSLOT * 12))  // NSLOT int
#define WS_XB       ((size_t)(128 * 1024))         // B*D bf16          = 4 MB
#define WS_HBUF     ((size_t)(8u  * 1024 * 1024))  // NSLOT*HDIM bf16   = 32 MB
#define WS_YBUF     ((size_t)(48u * 1024 * 1024))  // NSLOT*DMODEL f32  = 16 MB

// ---------------- GEMM tiling ----------------
#define MT        32                  // token-tile rows per block
#define KS        32                  // K per WMMA step (bf16 16x16x32)
#define NT_TILE   256                 // N columns per block (8 waves x 16x64)
#define NSUB      4                   // 16-wide N sub-tiles per wave
#define MAXTILES  (B_TOK / MT)        // worst case tiles per expert = 64

// ---------- async global->LDS copy of 8 bytes (4 bf16), guarded ----------
__device__ __forceinline__ void copy4bf_g2lds(const __bf16* __restrict__ g, __bf16* l) {
#if __has_builtin(__builtin_amdgcn_global_load_async_to_lds_b64)
    __builtin_amdgcn_global_load_async_to_lds_b64(
        (__attribute__((address_space(1))) v2i_gcc*)g,
        (__attribute__((address_space(3))) v2i_gcc*)l, 0, 0);
#else
    *(v4bf*)l = *(const v4bf*)g;
#endif
}
__device__ __forceinline__ void async_fence() {
#if __has_builtin(__builtin_amdgcn_global_load_async_to_lds_b64)
    asm volatile("s_wait_asynccnt 0x0" ::: "memory");
#endif
}

// =======================================================================
// 0. init: zero per-expert counters
// =======================================================================
__global__ void moe_init_kernel(int* counts, int* cursors) {
    int t = threadIdx.x;
    if (t < NEXP) { counts[t] = 0; cursors[t] = 0; }
}

// =======================================================================
// 1. gate: logits -> top-2 -> softmax; record selections, count per expert
// =======================================================================
__global__ __launch_bounds__(256) void moe_gate_kernel(
    const float* __restrict__ x, const float* __restrict__ gw,
    const float* __restrict__ gb, int* __restrict__ counts,
    int* __restrict__ sel_e, float* __restrict__ sel_g) {
    __shared__ float logits[NEXP];
    int b    = blockIdx.x;
    int e    = threadIdx.x >> 5;
    int lane = threadIdx.x & 31;
    const float* xr = x  + (size_t)b * DMODEL;
    const float* wr = gw + (size_t)e * DMODEL;
    float acc = 0.f;
    for (int i = lane; i < DMODEL; i += 32) acc = fmaf(xr[i], wr[i], acc);
    for (int m = 16; m; m >>= 1) acc += __shfl_xor(acc, m, 32);
    if (lane == 0) logits[e] = acc + gb[e];
    __syncthreads();
    if (threadIdx.x == 0) {
        float v0 = -INFINITY, v1 = -INFINITY; int i0 = 0, i1 = 0;
        for (int j = 0; j < NEXP; ++j) {
            float v = logits[j];
            if (v > v0)      { v1 = v0; i1 = i0; v0 = v; i0 = j; }
            else if (v > v1) { v1 = v;  i1 = j; }
        }
        float g0 = 1.0f, g1 = __expf(v1 - v0);
        float inv = 1.f / (g0 + g1);
        g0 *= inv; g1 *= inv;
        sel_e[b * 2 + 0] = i0; sel_e[b * 2 + 1] = i1;
        sel_g[b * 2 + 0] = g0; sel_g[b * 2 + 1] = g1;
        atomicAdd(&counts[i0], 1);
        atomicAdd(&counts[i1], 1);
    }
}

// =======================================================================
// 2. exclusive scan over 8 expert counts
// =======================================================================
__global__ void moe_scan_kernel(const int* __restrict__ counts, int* __restrict__ offsets) {
    if (threadIdx.x == 0) {
        int run = 0;
        for (int e = 0; e < NEXP; ++e) { offsets[e] = run; run += counts[e]; }
    }
}

// =======================================================================
// 3. dispatch: assign packed slot per (token, k)
// =======================================================================
__global__ __launch_bounds__(256) void moe_dispatch_kernel(
    const int* __restrict__ sel_e, const int* __restrict__ offsets,
    int* __restrict__ cursors, int* __restrict__ slot_of, int* __restrict__ token_of) {
    int b = blockIdx.x * 256 + threadIdx.x;
    if (b >= B_TOK) return;
    for (int k = 0; k < TOPK; ++k) {
        int e    = sel_e[b * 2 + k];
        int pos  = atomicAdd(&cursors[e], 1);
        int slot = offsets[e] + pos;
        slot_of[b * 2 + k] = slot;
        token_of[slot]     = b;
    }
}

// =======================================================================
// 4. convert activations fp32 -> bf16
// =======================================================================
__global__ __launch_bounds__(256) void moe_cvt_x_kernel(
    const float* __restrict__ x, __bf16* __restrict__ xb) {
    size_t i = ((size_t)blockIdx.x * 256 + threadIdx.x) * 4;
    v4f v = *(const v4f*)(x + i);
    v4bf o;
    o[0] = (__bf16)v[0]; o[1] = (__bf16)v[1];
    o[2] = (__bf16)v[2]; o[3] = (__bf16)v[3];
    *(v4bf*)(xb + i) = o;
}

// ---------- fragment concat ----------
__device__ __forceinline__ v16bf cat8(v8bf lo, v8bf hi) {
    return __builtin_shufflevector(lo, hi, 0,1,2,3,4,5,6,7,8,9,10,11,12,13,14,15);
}

// =======================================================================
// Shared WMMA GEMM body. Block tile: MT(32) x NT_TILE(256); 8 waves as
// 2(M) x 4(N); each wave: 16x64 via four v_wmma_f32_16x16x32_bf16 / K-step.
// =======================================================================
struct GemmFrag {
    v8f acc[NSUB];
};

__device__ __forceinline__ void gemm_loop(
    const __bf16* __restrict__ aptr,          // this thread's A source row
    const float* __restrict__ wbase,          // this thread's W source
    int ldb, int kdim,
    __bf16 (*As)[40], __bf16 (*Bs)[40],
    int arow, int acol, int bkr, int bcc,
    int lane, int wm, int wn, GemmFrag& f) {
#pragma unroll
    for (int s = 0; s < NSUB; ++s) f.acc[s] = (v8f){};
    for (int k0 = 0; k0 < kdim; k0 += KS) {
        // A tile: async global->LDS (bf16, 8B per thread, ASYNCcnt-tracked)
        copy4bf_g2lds(aptr + k0, &As[arow][acol]);
        // W tile: fp32 load, cvt to bf16, transposed scatter into Bs[N][K]
        const float* wp = wbase + (size_t)k0 * ldb;
        if (k0 + KS < kdim) __builtin_prefetch(wp + (size_t)KS * ldb, 0, 1);
#pragma unroll
        for (int j = 0; j < 8; ++j) {
            v4f wv = *(const v4f*)(wp + 4 * j);
#pragma unroll
            for (int i = 0; i < 4; ++i) Bs[bcc + 4 * j + i][bkr] = (__bf16)wv[i];
        }
        async_fence();
        __syncthreads();
        // A fragment per ISA 16-bit A 16x32 layout
        int am  = wm * 16 + (lane & 15);
        int klo = (lane < 16) ? 0 : 8;
        v16bf afrag = cat8(*(const v8bf*)&As[am][klo],
                           *(const v8bf*)&As[am][klo + 16]);
        // B fragments per ISA B 32x16 layout (same K order as A: klo..+7, klo+16..+23)
        int bkf = (lane < 16) ? 0 : 8;
        v16bf bfrag[NSUB];
#pragma unroll
        for (int s = 0; s < NSUB; ++s) {
            int bn = wn * 64 + s * 16 + (lane & 15);
            bfrag[s] = cat8(*(const v8bf*)&Bs[bn][bkf],
                            *(const v8bf*)&Bs[bn][bkf + 16]);
        }
#pragma unroll
        for (int s = 0; s < NSUB; ++s)
            f.acc[s] = __builtin_amdgcn_wmma_f32_16x16x32_bf16(
                false, afrag, false, bfrag[s], (short)0, f.acc[s], false, false);
        __syncthreads();
    }
}

// =======================================================================
// 5. GEMM1: h[slot,:] = gelu( x[token(slot),:] @ w1[e] + b1[e] ), bf16 out
// =======================================================================
__global__ __launch_bounds__(256) void moe_ffn1_kernel(
    const __bf16* __restrict__ xb, const float* __restrict__ w1,
    const float* __restrict__ b1, const int* __restrict__ counts,
    const int* __restrict__ offsets, const int* __restrict__ token_of,
    __bf16* __restrict__ hbuf) {
    const int NTN = HDIM / NT_TILE;                       // 16
    int bid = blockIdx.x;
    int e   = bid / (MAXTILES * NTN);
    int rem = bid % (MAXTILES * NTN);
    int mt  = rem / NTN;
    int nt  = rem % NTN;
    int cnt = counts[e];
    if (mt * MT >= cnt) return;                           // block-uniform exit
    int base = offsets[e];
    int n0   = nt * NT_TILE;

    __shared__ alignas(16) __bf16 As[MT][40];             // row-major [M][K]
    __shared__ alignas(16) __bf16 Bs[NT_TILE][40];        // transposed [N][K]

    int tid  = threadIdx.x;
    int lane = tid & 31;
    int wid  = tid >> 5;
    int wm   = wid >> 2;                                  // 0..1
    int wn   = wid & 3;                                   // 0..3

    int arow = tid >> 3;
    int acol = (tid & 7) * 4;
    int rloc = mt * MT + arow;
    int rcl  = rloc < cnt ? rloc : (cnt - 1);
    int tok  = token_of[base + rcl];
    const __bf16* aptr = xb + (size_t)tok * DMODEL + acol;

    int bkr = tid >> 3;
    int bcc = (tid & 7) * 32;
    const float* wbase = w1 + (size_t)e * DMODEL * HDIM + (size_t)bkr * HDIM + n0 + bcc;

    GemmFrag f;
    gemm_loop(aptr, wbase, HDIM, DMODEL, As, Bs, arow, acol, bkr, bcc, lane, wm, wn, f);

    int mhalf = (lane < 16) ? 0 : 8;                      // ISA C/D layout
#pragma unroll
    for (int s = 0; s < NSUB; ++s) {
        int nl   = wn * 64 + s * 16 + (lane & 15);
        float bv = b1[(size_t)e * HDIM + n0 + nl];
#pragma unroll
        for (int r = 0; r < 8; ++r) {
            int rowloc = mt * MT + wm * 16 + mhalf + r;
            if (rowloc < cnt) {
                size_t slot = (size_t)(base + rowloc);
                float v = f.acc[s][r] + bv;
                v = 0.5f * v * (1.0f + erff(v * 0.70710678118654752f));
                hbuf[slot * HDIM + n0 + nl] = (__bf16)v;
            }
        }
    }
}

// =======================================================================
// 6. GEMM2: y[slot,:] = h[slot,:] @ w2[e] + b2[e]  (fp32 out)
// =======================================================================
__global__ __launch_bounds__(256) void moe_ffn2_kernel(
    const __bf16* __restrict__ hbuf, const float* __restrict__ w2,
    const float* __restrict__ b2, const int* __restrict__ counts,
    const int* __restrict__ offsets, float* __restrict__ ybuf) {
    const int NTN = DMODEL / NT_TILE;                     // 4
    int bid = blockIdx.x;
    int e   = bid / (MAXTILES * NTN);
    int rem = bid % (MAXTILES * NTN);
    int mt  = rem / NTN;
    int nt  = rem % NTN;
    int cnt = counts[e];
    if (mt * MT >= cnt) return;
    int base = offsets[e];
    int n0   = nt * NT_TILE;

    __shared__ alignas(16) __bf16 As[MT][40];
    __shared__ alignas(16) __bf16 Bs[NT_TILE][40];

    int tid  = threadIdx.x;
    int lane = tid & 31;
    int wid  = tid >> 5;
    int wm   = wid >> 2;
    int wn   = wid & 3;

    int arow = tid >> 3;
    int acol = (tid & 7) * 4;
    int rloc = mt * MT + arow;
    int rcl  = rloc < cnt ? rloc : (cnt - 1);
    const __bf16* aptr = hbuf + (size_t)(base + rcl) * HDIM + acol;  // contiguous slots

    int bkr = tid >> 3;
    int bcc = (tid & 7) * 32;
    const float* wbase = w2 + (size_t)e * HDIM * DMODEL + (size_t)bkr * DMODEL + n0 + bcc;

    GemmFrag f;
    gemm_loop(aptr, wbase, DMODEL, HDIM, As, Bs, arow, acol, bkr, bcc, lane, wm, wn, f);

    int mhalf = (lane < 16) ? 0 : 8;
#pragma unroll
    for (int s = 0; s < NSUB; ++s) {
        int nl   = wn * 64 + s * 16 + (lane & 15);
        float bv = b2[(size_t)e * DMODEL + n0 + nl];
#pragma unroll
        for (int r = 0; r < 8; ++r) {
            int rowloc = mt * MT + wm * 16 + mhalf + r;
            if (rowloc < cnt) {
                size_t slot = (size_t)(base + rowloc);
                ybuf[slot * DMODEL + n0 + nl] = f.acc[s][r] + bv;
            }
        }
    }
}

// =======================================================================
// 7. combine: out[b,:] = g0*y[s0,:] + g1*y[s1,:]   (deterministic, no atomics)
// =======================================================================
__global__ __launch_bounds__(256) void moe_combine_kernel(
    const int* __restrict__ slot_of, const float* __restrict__ sel_g,
    const float* __restrict__ ybuf, float* __restrict__ out) {
    int b = blockIdx.x;
    int s0 = slot_of[b * 2], s1 = slot_of[b * 2 + 1];
    float g0 = sel_g[b * 2], g1 = sel_g[b * 2 + 1];
    const v4f* y0 = (const v4f*)(ybuf + (size_t)s0 * DMODEL);
    const v4f* y1 = (const v4f*)(ybuf + (size_t)s1 * DMODEL);
    v4f* o = (v4f*)(out + (size_t)b * DMODEL);
    for (int i = threadIdx.x; i < DMODEL / 4; i += 256)
        o[i] = g0 * y0[i] + g1 * y1[i];
}

// =======================================================================
extern "C" void kernel_launch(void* const* d_in, const int* in_sizes, int n_in,
                              void* d_out, int out_size, void* d_ws, size_t ws_size,
                              hipStream_t stream) {
    const float* x  = (const float*)d_in[0];
    const float* gw = (const float*)d_in[1];
    const float* gb = (const float*)d_in[2];
    const float* w1 = (const float*)d_in[3];
    const float* b1 = (const float*)d_in[4];
    const float* w2 = (const float*)d_in[5];
    const float* b2 = (const float*)d_in[6];
    float* out = (float*)d_out;

    char* ws = (char*)d_ws;
    int*    counts   = (int*)   (ws + WS_COUNTS);
    int*    cursors  = (int*)   (ws + WS_CURSORS);
    int*    offsets  = (int*)   (ws + WS_OFFSETS);
    int*    sel_e    = (int*)   (ws + WS_SEL_E);
    float*  sel_g    = (float*) (ws + WS_SEL_G);
    int*    slot_of  = (int*)   (ws + WS_SLOT_OF);
    int*    token_of = (int*)   (ws + WS_TOKEN_OF);
    __bf16* xb       = (__bf16*)(ws + WS_XB);
    __bf16* hbuf     = (__bf16*)(ws + WS_HBUF);
    float*  ybuf     = (float*) (ws + WS_YBUF);

    moe_init_kernel<<<1, 32, 0, stream>>>(counts, cursors);
    moe_gate_kernel<<<B_TOK, 256, 0, stream>>>(x, gw, gb, counts, sel_e, sel_g);
    moe_scan_kernel<<<1, 32, 0, stream>>>(counts, offsets);
    moe_dispatch_kernel<<<(B_TOK + 255) / 256, 256, 0, stream>>>(sel_e, offsets, cursors,
                                                                 slot_of, token_of);
    moe_cvt_x_kernel<<<(B_TOK * DMODEL) / (4 * 256), 256, 0, stream>>>(x, xb);
    moe_ffn1_kernel<<<NEXP * MAXTILES * (HDIM / NT_TILE), 256, 0, stream>>>(
        xb, w1, b1, counts, offsets, token_of, hbuf);
    moe_ffn2_kernel<<<NEXP * MAXTILES * (DMODEL / NT_TILE), 256, 0, stream>>>(
        hbuf, w2, b2, counts, offsets, ybuf);
    moe_combine_kernel<<<B_TOK, 256, 0, stream>>>(slot_of, sel_g, ybuf, out);
}
